// SelfAttention_3642132267110
// MI455X (gfx1250) — compile-verified, matrix operand order checked by reading, not run on hardware
//
#include <hip/hip_runtime.h>
#include <hip/hip_bf16.h>
#include <math.h>

// Problem constants (B=2, L=2048, DIM=2048, H=16, D=128)
#define BD   2
#define LQ   2048
#define DIMC 2048
#define NH   16
#define HD   128
#define N3   6144

typedef __bf16 bf16;
typedef __attribute__((ext_vector_type(16))) __bf16 bf16x16;
typedef __attribute__((ext_vector_type(8)))  __bf16 bf16x8;
typedef __attribute__((ext_vector_type(4)))  __bf16 bf16x4;
typedef __attribute__((ext_vector_type(8)))  float  f32x8;

// Build a 16-element bf16 WMMA fragment from two contiguous 16B LDS chunks.
// CDNA5 16-bit A/B layout: lanes 0-15 hold K=[0..7]+[16..23], lanes 16-31 hold
// K=[8..15]+[24..31] for their row/col -> two 8-element contiguous loads.
__device__ __forceinline__ bf16x16 frag2(const bf16* p0, const bf16* p1) {
  bf16x8 lo = *(const bf16x8*)p0;
  bf16x8 hi = *(const bf16x8*)p1;
  bf16x16 f;
#pragma unroll
  for (int i = 0; i < 8; ++i) { f[i] = lo[i]; f[i + 8] = hi[i]; }
  return f;
}

#define WMMA_BF16(A_, B_, C_) \
  __builtin_amdgcn_wmma_f32_16x16x32_bf16(false, (A_), false, (B_), (short)0, (C_), false, false)

// CDNA5 async global->LDS copy: 16B per lane, tracked by ASYNCcnt, no VGPR
// data staging (CDNA5 ISA).  LDS address is taken from the vdst VGPR.
__device__ __forceinline__ void async_b128(bf16* lds_dst, const bf16* gsrc) {
  asm volatile("global_load_async_to_lds_b128 %0, %1, off"
               :
               : "v"((unsigned)(size_t)lds_dst), "v"((unsigned long long)(size_t)gsrc)
               : "memory");
}
__device__ __forceinline__ void wait_async0() {
  asm volatile("s_wait_asynccnt 0x0" ::: "memory");
}

// ---------------------------------------------------------------------------
// fp32 -> bf16 bulk convert (vectorized 4-wide)
// ---------------------------------------------------------------------------
__global__ void cvt_bf16(const float* __restrict__ in, bf16* __restrict__ out, int n4) {
  int i = blockIdx.x * blockDim.x + threadIdx.x;
  if (i >= n4) return;
  float4 v = ((const float4*)in)[i];
  bf16x4 o;
  o[0] = (bf16)v.x; o[1] = (bf16)v.y; o[2] = (bf16)v.z; o[3] = (bf16)v.w;
  ((bf16x4*)out)[i] = o;
}

// ---------------------------------------------------------------------------
// fp32 [rows][cols] -> bf16 transposed [cols][rows], 32x32 LDS tiles.
// Makes GEMM B-tiles contiguous so the mainloop can use async-to-LDS copies.
// ---------------------------------------------------------------------------
__global__ void cvt_transpose(const float* __restrict__ in, bf16* __restrict__ out,
                              int rows, int cols) {
  __shared__ float tile[32][33];
  const int c0 = blockIdx.x * 32, r0 = blockIdx.y * 32;
  const int tx = threadIdx.x & 31, ty = threadIdx.x >> 5;  // 256 thr: ty 0..7
#pragma unroll
  for (int i = 0; i < 32; i += 8)
    tile[ty + i][tx] = in[(size_t)(r0 + ty + i) * cols + (c0 + tx)];
  __syncthreads();
#pragma unroll
  for (int i = 0; i < 32; i += 8)
    out[(size_t)(c0 + ty + i) * rows + (r0 + tx)] = (bf16)tile[tx][ty + i];
}

// ---------------------------------------------------------------------------
// 128x128-tile bf16 WMMA GEMM, fp32 accumulate, double-buffered async staging.
// Buffers are addressed only through direct expressions on the shared-memory
// symbol (no pointer arrays) so fragment loads lower to ds_load_b128.
// B operand is pre-transposed Wt[N][K], so both tiles are contiguous rows.
// MODE 0: QKV epilogue (bias + per-head RMSNorm + RoPE; q/k -> [B,H,L,D],
//         v -> transposed [B,H,D,L] so attention can async-copy V^T tiles).
// MODE 1: proj epilogue (bias, fp32 store to d_out).
// Block: 256 threads = 8 wave32; wave grid 4(M)x2(N); wave tile 32x64.
// ---------------------------------------------------------------------------
template <int MODE>
__global__ void __launch_bounds__(256, 1)
gemm128(const bf16* __restrict__ A, const bf16* __restrict__ Wt,
        const float* __restrict__ bias,
        const float* __restrict__ q_scale, const float* __restrict__ k_scale,
        const float* __restrict__ pe,
        bf16* __restrict__ qo, bf16* __restrict__ ko, bf16* __restrict__ vo,
        float* __restrict__ fo, int N, int K) {
  const int KC = 64, LDA = 72, LDC = 132;  // padded LDS strides
  extern __shared__ char smem[];
  bf16* sA0 = (bf16*)smem;
  bf16* sB0 = sA0 + 128 * LDA;
  bf16* sA1 = sB0 + 128 * LDA;
  bf16* sB1 = sA1 + 128 * LDA;
  float* sC = (float*)(sB1 + 128 * LDA);  // [128][LDC] accumulator spill

  const int t = threadIdx.x;
  const int wave = t >> 5, lane = t & 31;
  const int lrow = lane & 15, lhalf = (lane >> 4) << 3;
  const int wM = (wave >> 1) * 32, wN = (wave & 1) * 64;
  const int m0 = blockIdx.y * 128, n0 = blockIdx.x * 128;

  // per-thread staging chunk coordinates (16B granules, 128x64 tile)
  const int sr = (t * 8) >> 6, scc = (t * 8) & 63;  // j-th chunk adds 32 rows

  f32x8 acc[2][4];
#pragma unroll
  for (int i = 0; i < 2; ++i)
#pragma unroll
    for (int j = 0; j < 4; ++j)
#pragma unroll
      for (int r = 0; r < 8; ++r) acc[i][j][r] = 0.0f;

  auto stage = [&](bf16* dA, bf16* dB, int k0) {
#pragma unroll
    for (int j = 0; j < 4; ++j) {
      async_b128(dA + (sr + j * 32) * LDA + scc,
                 A + (size_t)(m0 + sr + j * 32) * K + k0 + scc);
      async_b128(dB + (sr + j * 32) * LDA + scc,
                 Wt + (size_t)(n0 + sr + j * 32) * K + k0 + scc);
    }
  };
  auto compute = [&](const bf16* cA, const bf16* cB) {
#pragma unroll
    for (int kk = 0; kk < KC; kk += 32) {
      bf16x16 af[2], bfr[4];
#pragma unroll
      for (int i = 0; i < 2; ++i) {
        const bf16* p = cA + (wM + i * 16 + lrow) * LDA + kk + lhalf;
        af[i] = frag2(p, p + 16);
      }
#pragma unroll
      for (int j = 0; j < 4; ++j) {
        const bf16* p = cB + (wN + j * 16 + lrow) * LDA + kk + lhalf;
        bfr[j] = frag2(p, p + 16);
      }
#pragma unroll
      for (int i = 0; i < 2; ++i)
#pragma unroll
        for (int j = 0; j < 4; ++j)
          acc[i][j] = WMMA_BF16(af[i], bfr[j], acc[i][j]);
    }
  };

  stage(sA0, sB0, 0);
  // K is a multiple of 2*KC (=128) for all uses here.
  for (int k0 = 0; k0 < K; k0 += 2 * KC) {
    wait_async0();
    __syncthreads();
    stage(sA1, sB1, k0 + KC);  // k0+KC < K always (K multiple of 2*KC)
    compute(sA0, sB0);
    wait_async0();
    __syncthreads();
    if (k0 + 2 * KC < K) stage(sA0, sB0, k0 + 2 * KC);
    compute(sA1, sB1);
  }
  __syncthreads();

  // Spill accumulators (C layout: row = r + lane_half*8, col = lane%16)
#pragma unroll
  for (int i = 0; i < 2; ++i)
#pragma unroll
    for (int j = 0; j < 4; ++j)
#pragma unroll
      for (int r = 0; r < 8; ++r)
        sC[(wM + i * 16 + lhalf + r) * LDC + (wN + j * 16 + lrow)] = acc[i][j][r];
  __syncthreads();

  if (MODE == 1) {
    for (int j = 0; j < 64; ++j) {
      int e = j * 256 + t;
      int r = e >> 7, c = e & 127;
      fo[(size_t)(m0 + r) * N + (n0 + c)] = sC[r * LDC + c] + bias[n0 + c];
    }
  } else {
    if (t < 128) {
      const int r = t;
      const int mtype = n0 >> 11;          // 0=q 1=k 2=v
      const int head  = (n0 & 2047) >> 7;
      const int grow = m0 + r;
      const int b = grow >> 11, l = grow & 2047;
      float* row = sC + r * LDC;
      const float* bia = bias + n0;
      if (mtype == 2) {
        // V stored transposed: vo[(b*NH+h)*HD + d][l]
        bf16* op = vo + (size_t)(b * NH + head) * HD * LQ + l;
        for (int c = 0; c < 128; ++c)
          op[(size_t)c * LQ] = (bf16)(row[c] + bia[c]);
      } else {
        float ss = 0.f;
        for (int c = 0; c < 128; ++c) {
          float x = row[c] + bia[c];
          row[c] = x;
          ss += x * x;
        }
        float rr = rsqrtf(ss * (1.0f / 128.0f) + 1e-6f);
        const float* sc = (mtype == 0) ? q_scale : k_scale;
        bf16* op = ((mtype == 0) ? qo : ko) + ((size_t)(b * NH + head) * LQ + l) * HD;
        const float* per = pe + (size_t)l * 256;  // [l][64][2][2]
        for (int d2 = 0; d2 < 64; ++d2) {
          float x0 = row[2 * d2]     * rr * sc[2 * d2];
          float x1 = row[2 * d2 + 1] * rr * sc[2 * d2 + 1];
          float c00 = per[4 * d2],     c01 = per[4 * d2 + 1];
          float c10 = per[4 * d2 + 2], c11 = per[4 * d2 + 3];
          op[2 * d2]     = (bf16)(c00 * x0 + c01 * x1);
          op[2 * d2 + 1] = (bf16)(c10 * x0 + c11 * x1);
        }
      }
    }
  }
}

// ---------------------------------------------------------------------------
// Flash-attention tile kernel: one block = 128 queries of one (b,h);
// 16 key tiles of 128, double-buffered async K/V^T staging (unrolled x2 so
// all LDS pointers are direct expressions -> ds_load lowering).
// V is pre-transposed [bh][d][l] so V^T tiles are contiguous-row async copies.
// LDS ~272 KB (CDNA5 320 KB WGP LDS); launch_bounds(256,1) grants the ~200
// VGPRs/lane needed to keep S and O accumulators resident (no scratch spill).
// ---------------------------------------------------------------------------
__global__ void __launch_bounds__(256, 1)
attn128(const bf16* __restrict__ Q, const bf16* __restrict__ Kv,
        const bf16* __restrict__ Vt, bf16* __restrict__ O) {
  const int LDB = 136, LDSS = 132;
  extern __shared__ char smem[];
  bf16* sQ   = (bf16*)smem;            // [128][136] queries
  bf16* sK0  = sQ + 128 * LDB;
  bf16* sK1  = sK0 + 128 * LDB;
  bf16* sVt0 = sK1 + 128 * LDB;        // [d][key]
  bf16* sVt1 = sVt0 + 128 * LDB;
  float* sS  = (float*)(sVt1 + 128 * LDB);  // [128][132] scores fp32
  bf16*  sP  = (bf16*)(sS + 128 * LDSS);    // [128][136] probs bf16
  float* sM  = (float*)(sP + 128 * LDB);
  float* sL  = sM + 128;
  float* sAl = sL + 128;

  const int t = threadIdx.x;
  const int wave = t >> 5, lane = t & 31;
  const int lrow = lane & 15, lhalf = (lane >> 4) << 3;
  const int bh = blockIdx.y;
  const int q0 = blockIdx.x * 128;
  const size_t base  = (size_t)bh * LQ * HD;  // q/k rows [l][d]
  const size_t baseT = (size_t)bh * HD * LQ;  // v^T rows [d][l]
  const int qrow = wave * 16;
  const int NT = LQ / 128;

  // per-thread 16B staging chunk within a 128x128 tile
  const int sr = (t * 8) >> 7, scc = (t * 8) & 127;  // j-th chunk adds 16 rows

  auto stageKV = [&](bf16* dK, bf16* dV, int kn) {
#pragma unroll
    for (int j = 0; j < 8; ++j) {
      const int r = sr + j * 16;
      async_b128(dK + r * LDB + scc, Kv + base + (size_t)(kn + r) * HD + scc);
      async_b128(dV + r * LDB + scc, Vt + baseT + (size_t)r * LQ + kn + scc);
    }
  };

  f32x8 oacc[8];
#pragma unroll
  for (int j = 0; j < 8; ++j)
#pragma unroll
    for (int r = 0; r < 8; ++r) oacc[j][r] = 0.f;

  auto tile = [&](const bf16* cK, const bf16* cV, bf16* nK, bf16* nV, int kt) {
    wait_async0();
    __syncthreads();
    if (kt + 1 < NT) stageKV(nK, nV, (kt + 1) * 128);

    // S(16q x 128k) per wave: contraction over d=128 in 4 WMMA k-steps
    f32x8 s[8];
#pragma unroll
    for (int j = 0; j < 8; ++j)
#pragma unroll
      for (int r = 0; r < 8; ++r) s[j][r] = 0.f;
#pragma unroll
    for (int dk = 0; dk < 128; dk += 32) {
      const bf16* pa = sQ + (qrow + lrow) * LDB + dk + lhalf;
      bf16x16 af = frag2(pa, pa + 16);
#pragma unroll
      for (int j = 0; j < 8; ++j) {
        const bf16* pb = cK + (j * 16 + lrow) * LDB + dk + lhalf;
        bf16x16 bb = frag2(pb, pb + 16);
        s[j] = WMMA_BF16(af, bb, s[j]);
      }
    }
    const float scl = 0.088388347648318447f;  // 1/sqrt(128)
#pragma unroll
    for (int j = 0; j < 8; ++j)
#pragma unroll
      for (int r = 0; r < 8; ++r)
        sS[(qrow + lhalf + r) * LDSS + (j * 16 + lrow)] = s[j][r] * scl;
    __syncthreads();

    // Online softmax, one thread per query row
    if (t < 128) {
      float mo = sM[t], mx = mo;
      const float* srow = sS + t * LDSS;
      for (int c = 0; c < 128; ++c) mx = fmaxf(mx, srow[c]);
      float al = __expf(mo - mx);
      float ls = sL[t] * al;
      bf16* pr = sP + t * LDB;
      for (int c = 0; c < 128; ++c) {
        float p = __expf(srow[c] - mx);
        ls += p;
        pr[c] = (bf16)p;
      }
      sM[t] = mx; sL[t] = ls; sAl[t] = al;
    }
    __syncthreads();

    // Rescale running O, accumulate P @ V
    float al[8];
#pragma unroll
    for (int r = 0; r < 8; ++r) al[r] = sAl[qrow + lhalf + r];
#pragma unroll
    for (int j = 0; j < 8; ++j)
#pragma unroll
      for (int r = 0; r < 8; ++r) oacc[j][r] *= al[r];

#pragma unroll
    for (int kk = 0; kk < 128; kk += 32) {
      const bf16* pa = sP + (qrow + lrow) * LDB + kk + lhalf;
      bf16x16 af = frag2(pa, pa + 16);
#pragma unroll
      for (int j = 0; j < 8; ++j) {
        const bf16* pb = cV + (j * 16 + lrow) * LDB + kk + lhalf;
        bf16x16 bb = frag2(pb, pb + 16);
        oacc[j] = WMMA_BF16(af, bb, oacc[j]);
      }
    }
  };

  // prologue: async-stage Q tile and (K,V^T) tile 0
#pragma unroll
  for (int j = 0; j < 8; ++j) {
    const int r = sr + j * 16;
    async_b128(sQ + r * LDB + scc, Q + base + (size_t)(q0 + r) * HD + scc);
  }
  stageKV(sK0, sVt0, 0);
  if (t < 128) { sM[t] = -1e30f; sL[t] = 0.f; }

  // NT is even: ping-pong with compile-time buffer pointers
  for (int kt = 0; kt < NT; kt += 2) {
    tile(sK0, sVt0, sK1, sVt1, kt);
    tile(sK1, sVt1, sK0, sVt0, kt + 1);
  }
  __syncthreads();

  // Normalize and store O bf16 in [B, L, H*D] (row layout for proj GEMM)
  float linv[8];
#pragma unroll
  for (int r = 0; r < 8; ++r) linv[r] = 1.0f / sL[qrow + lhalf + r];
  const int b = bh >> 4, h = bh & 15;
#pragma unroll
  for (int j = 0; j < 8; ++j) {
    int d = j * 16 + lrow;
#pragma unroll
    for (int r = 0; r < 8; ++r) {
      int l = q0 + qrow + lhalf + r;
      O[((size_t)(b * LQ + l)) * DIMC + h * HD + d] = (bf16)(oacc[j][r] * linv[r]);
    }
  }
}

// ---------------------------------------------------------------------------
extern "C" void kernel_launch(void* const* d_in, const int* in_sizes, int n_in,
                              void* d_out, int out_size, void* d_ws, size_t ws_size,
                              hipStream_t stream) {
  const float* x       = (const float*)d_in[0];
  const float* pe      = (const float*)d_in[1];
  const float* qkv_w   = (const float*)d_in[2];
  const float* qkv_b   = (const float*)d_in[3];
  const float* q_scale = (const float*)d_in[4];
  const float* k_scale = (const float*)d_in[5];
  const float* proj_w  = (const float*)d_in[6];
  const float* proj_b  = (const float*)d_in[7];
  float* out = (float*)d_out;

  char* ws = (char*)d_ws;
  size_t off = 0;
  auto alloc = [&](size_t bytes) -> char* {
    char* p = ws + off;
    off += (bytes + 255) & ~(size_t)255;
    return p;
  };
  bf16* x_bf  = (bf16*)alloc((size_t)BD * LQ * DIMC * 2);     // 16.8 MB
  bf16* qwT   = (bf16*)alloc((size_t)DIMC * N3 * 2);          // 25.2 MB  [6144][2048]
  bf16* pwT   = (bf16*)alloc((size_t)DIMC * DIMC * 2);        //  8.4 MB  [2048][2048]
  bf16* q_bf  = (bf16*)alloc((size_t)BD * NH * LQ * HD * 2);  // 16.8 MB  [bh][l][d]
  bf16* k_bf  = (bf16*)alloc((size_t)BD * NH * LQ * HD * 2);  //          [bh][l][d]
  bf16* vT_bf = (bf16*)alloc((size_t)BD * NH * LQ * HD * 2);  //          [bh][d][l]
  bf16* o_bf  = (bf16*)alloc((size_t)BD * LQ * DIMC * 2);
  if (off > ws_size) return;  // workspace too small (deterministic no-op)

  int n4 = BD * LQ * DIMC / 4;
  cvt_bf16<<<(n4 + 255) / 256, 256, 0, stream>>>(x, x_bf, n4);
  cvt_transpose<<<dim3(N3 / 32, DIMC / 32), 256, 0, stream>>>(qkv_w, qwT, DIMC, N3);
  cvt_transpose<<<dim3(DIMC / 32, DIMC / 32), 256, 0, stream>>>(proj_w, pwT, DIMC, DIMC);

  const size_t gemm_smem = (size_t)4 * 128 * 72 * 2 + (size_t)128 * 132 * 4;  // 141312
  const size_t attn_smem = (size_t)5 * 128 * 136 * 2 + (size_t)128 * 132 * 4 +
                           (size_t)128 * 136 * 2 + 3 * 128 * 4;               // 278016

  dim3 g1(N3 / 128, (BD * LQ) / 128);
  gemm128<0><<<g1, 256, gemm_smem, stream>>>(x_bf, qwT, qkv_b, q_scale, k_scale,
                                             pe, q_bf, k_bf, vT_bf, nullptr, N3, DIMC);

  dim3 g2(LQ / 128, BD * NH);
  attn128<<<g2, 256, attn_smem, stream>>>(q_bf, k_bf, vT_bf, o_bf);

  dim3 g3(DIMC / 128, (BD * LQ) / 128);
  gemm128<1><<<g3, 256, gemm_smem, stream>>>(o_bf, pwT, proj_b, nullptr, nullptr,
                                             nullptr, nullptr, nullptr, nullptr,
                                             out, DIMC, DIMC);
}
